// Model_33998961115361
// MI455X (gfx1250) — matrix operand order repaired and checked by reading, not compiled, not run on hardware
//
#include <hip/hip_runtime.h>
#include <hip/hip_bf16.h>

// ---------------------------------------------------------------------------
// MI455X (gfx1250) dense VGG-ish 3D CNN.
//   * inter-layer activations stored as f16 NDHWC (halves HBM traffic)
//   * every conv / linear = implicit GEMM on V_WMMA_F32_16X16X32_F16
//   * weights pre-packed on device into the exact WMMA B-fragment layout
//   * register tiling: each wave holds MACC x NACC accumulator tiles,
//     reusing A fragments across NACC couts and B fragments across MACC
//     voxel tiles (MACC+NACC loads per MACC*NACC WMMAs)
//   * BN+ReLU fused into the WMMA epilogue
// ---------------------------------------------------------------------------

typedef __attribute__((ext_vector_type(16))) _Float16 v16h;
typedef __attribute__((ext_vector_type(8)))  _Float16 v8h;
typedef __attribute__((ext_vector_type(8)))  float    v8f;

// ---------------------------------------------------------------------------
// pack fp32 input [vox][CIN] -> f16 [vox][CS] with zero channel padding
// ---------------------------------------------------------------------------
__global__ __launch_bounds__(256) void pack_in(const float* __restrict__ x,
                                               _Float16* __restrict__ out,
                                               int CIN, int CS, long long total) {
  long long idx = (long long)blockIdx.x * blockDim.x + threadIdx.x;
  if (idx >= total) return;
  int c = (int)(idx % CS);
  long long v = idx / CS;
  out[idx] = (_Float16)((c < CIN) ? x[v * CIN + c] : 0.0f);
}

// ---------------------------------------------------------------------------
// pack fp32 weights [TAPS][CIN][COUT] -> f16 WMMA-B fragments
// layout: [COUT/16][TAPS][KCH][lane=32][j=16], element (lane,j):
//   k = kc*32 + (j>>3)*16 + (lane>>4)*8 + (j&7)   (zero when k >= CIN)
//   n = ctile*16 + (lane&15)
// ---------------------------------------------------------------------------
__global__ __launch_bounds__(256) void pack_w(const float* __restrict__ w,
                                              _Float16* __restrict__ wp,
                                              int CIN, int COUT, int TAPS, int KCH,
                                              int total) {
  int idx = blockIdx.x * blockDim.x + threadIdx.x;
  if (idx >= total) return;
  int j = idx & 15;
  int lane = (idx >> 4) & 31;
  int t = idx >> 9;
  int kc = t % KCH; t /= KCH;
  int tap = t % TAPS;
  int ct = t / TAPS;
  int k = kc * 32 + ((j >> 3) << 4) + ((lane >> 4) << 3) + (j & 7);
  int n = ct * 16 + (lane & 15);
  float v = (k < CIN) ? w[((size_t)tap * CIN + k) * COUT + n] : 0.0f;
  wp[idx] = (_Float16)v;
}

// ---------------------------------------------------------------------------
// maxpool 3x3x3 stride 2 VALID, f16 NDHWC
// ---------------------------------------------------------------------------
__global__ __launch_bounds__(256) void maxpool3(const _Float16* __restrict__ in,
                                                _Float16* __restrict__ out,
                                                int SIN, int SOUT, int C, int total) {
  int idx = blockIdx.x * blockDim.x + threadIdx.x;
  if (idx >= total) return;
  int c = idx % C; int t = idx / C;
  int x = t % SOUT; t /= SOUT;
  int y = t % SOUT; t /= SOUT;
  int z = t % SOUT; int b = t / SOUT;
  float m = -3.0e38f;
  for (int dz = 0; dz < 3; ++dz)
    for (int dy = 0; dy < 3; ++dy)
      for (int dx = 0; dx < 3; ++dx) {
        size_t ii = (((size_t)(b * SIN + z * 2 + dz) * SIN + (y * 2 + dy)) * SIN
                     + (x * 2 + dx)) * C + c;
        float v = (float)in[ii];
        m = v > m ? v : m;
      }
  out[idx] = (_Float16)m;
}

// ---------------------------------------------------------------------------
// implicit-GEMM conv (+ fused BN/ReLU or bias) on WMMA f32_16x16x32_f16,
// register-tiled MACC (voxel tiles) x NACC (cout tiles) per wave.
// ---------------------------------------------------------------------------
template <int CINS, int COUT, int KS, int SIN, int SOUT, int STRIDE, int PAD,
          int MACC, int NACC, bool LIN>
__global__ __launch_bounds__(256) void conv_wmma(
    const _Float16* __restrict__ in, const _Float16* __restrict__ wpk,
    const float* __restrict__ e0, const float* __restrict__ e1,
    const float* __restrict__ e2, const float* __restrict__ e3,
    void* __restrict__ outv, int batch) {
  constexpr int KCH = (CINS + 31) / 32;
  constexpr int NT = COUT / 16;
  constexpr int NG = NT / NACC;
  constexpr int TAPS = KS * KS * KS;
  static_assert(NT % NACC == 0, "NACC must divide COUT/16");

  const int totalM = batch * SOUT * SOUT * SOUT;
  const int mtiles = (totalM + 15) >> 4;
  const int mgroups = (mtiles + MACC - 1) / MACC;
  const int wave = (int)((blockIdx.x * blockDim.x + threadIdx.x) >> 5);
  const int lane = (int)(threadIdx.x & 31);
  const int cg = wave % NG;
  const int mg = wave / NG;
  if (mg >= mgroups) return;  // wave-uniform exit; EXEC stays all-ones

  const int half = lane >> 4;
  const int lm = lane & 15;

  int x[MACC], y[MACC], z[MACC], b[MACC];
  bool mv[MACC];
#pragma unroll
  for (int mi = 0; mi < MACC; ++mi) {
    const int midx = (mg * MACC + mi) * 16 + lm;
    mv[mi] = midx < totalM;
    int t = mv[mi] ? midx : 0;
    x[mi] = t % SOUT; t /= SOUT;
    y[mi] = t % SOUT; t /= SOUT;
    z[mi] = t % SOUT; b[mi] = t / SOUT;
  }

  v8f acc[MACC][NACC];
#pragma unroll
  for (int mi = 0; mi < MACC; ++mi)
#pragma unroll
    for (int ni = 0; ni < NACC; ++ni) {
      v8f zz = {};
      acc[mi][ni] = zz;
    }

  const _Float16* wlane = wpk + (size_t)lane * 16;

  for (int kd = 0; kd < KS; ++kd)
    for (int kh = 0; kh < KS; ++kh)
      for (int kw = 0; kw < KS; ++kw) {
        const int tap = (kd * KS + kh) * KS + kw;
        const _Float16* vox[MACC];
        bool inb[MACC];
#pragma unroll
        for (int mi = 0; mi < MACC; ++mi) {
          const int zi = z[mi] * STRIDE + kd - PAD;
          const int yi = y[mi] * STRIDE + kh - PAD;
          const int xi = x[mi] * STRIDE + kw - PAD;
          inb[mi] = mv[mi] && ((unsigned)zi < (unsigned)SIN) &&
                    ((unsigned)yi < (unsigned)SIN) &&
                    ((unsigned)xi < (unsigned)SIN);
          vox[mi] = in +
                    (size_t)(((b[mi] * SIN + zi) * SIN + yi) * SIN + xi) * CINS +
                    half * 8;
        }
#pragma unroll
        for (int kc = 0; kc < KCH; ++kc) {
          v16h afrag[MACC];
#pragma unroll
          for (int mi = 0; mi < MACC; ++mi) {
            v16h a = {};
            if (inb[mi]) {  // per-lane zero fill implements SAME-padding
              const v8h a0 = *(const v8h*)(vox[mi] + kc * 32);
              const v8h a1 = *(const v8h*)(vox[mi] + kc * 32 + 16);
#pragma unroll
              for (int i = 0; i < 8; ++i) { a[i] = a0[i]; a[i + 8] = a1[i]; }
            }
            afrag[mi] = a;
          }
#pragma unroll
          for (int ni = 0; ni < NACC; ++ni) {
            const int ct = cg * NACC + ni;
            const v16h bfrag = *(const v16h*)(
                wlane + ((size_t)ct * TAPS + tap) * KCH * 512 + (size_t)kc * 512);
#pragma unroll
            for (int mi = 0; mi < MACC; ++mi)
              acc[mi][ni] = __builtin_amdgcn_wmma_f32_16x16x32_f16(
                  false, afrag[mi], false, bfrag, (short)0, acc[mi][ni], false,
                  false);
          }
        }
      }

  // epilogue: C/D layout -> lane L, VGPR e holds (M = half*8 + e, N = L%16)
#pragma unroll
  for (int ni = 0; ni < NACC; ++ni) {
    const int n = (cg * NACC + ni) * 16 + lm;
    if (!LIN) {
      const float sc = e0[n] * __frsqrt_rn(e3[n] + 1e-4f);
      const float sh = e1[n] - e2[n] * sc;
      _Float16* out = (_Float16*)outv;
#pragma unroll
      for (int mi = 0; mi < MACC; ++mi)
#pragma unroll
        for (int e = 0; e < 8; ++e) {
          const int m = (mg * MACC + mi) * 16 + half * 8 + e;
          if (m < totalM) {
            float v = acc[mi][ni][e] * sc + sh;
            out[(size_t)m * COUT + n] = (_Float16)(v > 0.0f ? v : 0.0f);
          }
        }
    } else {
      const float bias = e0[n];
      float* out = (float*)outv;
#pragma unroll
      for (int mi = 0; mi < MACC; ++mi)
#pragma unroll
        for (int e = 0; e < 8; ++e) {
          const int m = (mg * MACC + mi) * 16 + half * 8 + e;
          if (m < totalM) out[(size_t)m * COUT + n] = acc[mi][ni][e] + bias;
        }
    }
  }
}

// ---------------------------------------------------------------------------
// host orchestration
// ---------------------------------------------------------------------------
static inline unsigned cdiv_ll(long long a, int b) {
  return (unsigned)((a + b - 1) / b);
}

extern "C" void kernel_launch(void* const* d_in, const int* in_sizes, int n_in,
                              void* d_out, int out_size, void* d_ws,
                              size_t ws_size, hipStream_t stream) {
  (void)in_sizes; (void)n_in; (void)out_size; (void)ws_size;
  // d_in layout (setup_inputs dict order, nested dicts flattened in insertion
  // order): 0:x, 1..10:convs[0..9], 11..50: bns[i] as (g,b,m,v),
  // 51:final_w, 52..55:final_bn (g,b,m,v), 56:lin_w, 57:lin_b
  const float* x = (const float*)d_in[0];

  char* ws = (char*)d_ws;
  _Float16* A = (_Float16*)ws;                              // 128 MiB region
  _Float16* B = (_Float16*)(ws + (size_t)134217728);        // 128 MiB region
  _Float16* wbase = (_Float16*)(ws + (size_t)268435456);    // packed weights

  // ---- pack weights (10 convs + final conv + linear) ----
  const int CINv[12]  = {3, 16, 16, 32, 32, 48, 48, 64, 64, 96, 96, 128};
  const int COUTv[12] = {16, 16, 32, 32, 48, 48, 64, 64, 96, 96, 128, 256};
  const int KCHv[12]  = {1, 1, 1, 1, 1, 2, 2, 2, 2, 3, 3, 4};  // ceil(CINS/32)
  const int TAPSv[12] = {27, 27, 27, 27, 27, 27, 27, 27, 27, 27, 27, 1};
  const float* wsrc[12];
  for (int i = 0; i < 10; ++i) wsrc[i] = (const float*)d_in[1 + i];
  wsrc[10] = (const float*)d_in[51];
  wsrc[11] = (const float*)d_in[56];

  _Float16* wp[12];
  size_t off = 0;
  for (int i = 0; i < 12; ++i) {
    wp[i] = wbase + off;
    off += (size_t)(COUTv[i] / 16) * TAPSv[i] * KCHv[i] * 512;
  }
  for (int i = 0; i < 12; ++i) {
    int total = (COUTv[i] / 16) * TAPSv[i] * KCHv[i] * 512;
    pack_w<<<dim3(cdiv_ll(total, 256)), dim3(256), 0, stream>>>(
        wsrc[i], wp[i], CINv[i], COUTv[i], TAPSv[i], KCHv[i], total);
  }

  // ---- pack input: fp32 [16,63^3,3] -> f16 [16,63^3,16] ----
  {
    long long total = (long long)16 * 63 * 63 * 63 * 16;
    pack_in<<<dim3(cdiv_ll(total, 256)), dim3(256), 0, stream>>>(x, A, 3, 16,
                                                                 total);
  }

#define RUN_CONV(CINS, COUT, SIN, SOUT, MACC, NACC, INP, WIDX, BNI, OUTP)     \
  do {                                                                        \
    const int mt_ = (16 * (SOUT) * (SOUT) * (SOUT) + 15) / 16;                \
    const int tiles_ =                                                        \
        ((mt_ + (MACC) - 1) / (MACC)) * (((COUT) / 16) / (NACC));             \
    conv_wmma<CINS, COUT, 3, SIN, SOUT, 1, 1, MACC, NACC, false>              \
        <<<dim3((tiles_ + 7) / 8), dim3(256), 0, stream>>>(                   \
            INP, wp[WIDX], (const float*)d_in[11 + 4 * (BNI)],                \
            (const float*)d_in[12 + 4 * (BNI)],                               \
            (const float*)d_in[13 + 4 * (BNI)],                               \
            (const float*)d_in[14 + 4 * (BNI)], (void*)(OUTP), 16);           \
  } while (0)

#define RUN_POOL(SIN, SOUT, C, INP, OUTP)                                     \
  do {                                                                        \
    const int total_ = 16 * (SOUT) * (SOUT) * (SOUT) * (C);                   \
    maxpool3<<<dim3(cdiv_ll(total_, 256)), dim3(256), 0, stream>>>(           \
        INP, OUTP, SIN, SOUT, C, total_);                                     \
  } while (0)

  RUN_CONV(16, 16, 63, 63, 2, 1, A, 0, 0, B);   // conv0 (Cin 3 pad->16)
  RUN_CONV(16, 16, 63, 63, 2, 1, B, 1, 1, A);   // conv1
  RUN_POOL(63, 31, 16, A, B);                   // pool 63->31
  RUN_CONV(16, 32, 31, 31, 2, 2, B, 2, 2, A);   // conv2
  RUN_CONV(32, 32, 31, 31, 2, 2, A, 3, 3, B);   // conv3
  RUN_POOL(31, 15, 32, B, A);                   // pool 31->15
  RUN_CONV(32, 48, 15, 15, 2, 3, A, 4, 4, B);   // conv4
  RUN_CONV(48, 48, 15, 15, 2, 3, B, 5, 5, A);   // conv5
  RUN_POOL(15, 7, 48, A, B);                    // pool 15->7
  RUN_CONV(48, 64, 7, 7, 2, 2, B, 6, 6, A);     // conv6
  RUN_CONV(64, 64, 7, 7, 2, 2, A, 7, 7, B);     // conv7
  RUN_POOL(7, 3, 64, B, A);                     // pool 7->3
  RUN_CONV(64, 96, 3, 3, 1, 3, A, 8, 8, B);     // conv8
  RUN_CONV(96, 96, 3, 3, 1, 3, B, 9, 9, A);     // conv9

  // final conv: 3^3 stride-2 VALID -> 1 voxel, 96->128, BN+ReLU, f16 out in B
  {
    const int tiles_ = 1 * (128 / 16);  // mtiles=1 (M=16 batch), 8 ctiles
    conv_wmma<96, 128, 3, 3, 1, 2, 0, 1, 1, false>
        <<<dim3((tiles_ + 7) / 8), dim3(256), 0, stream>>>(
            A, wp[10], (const float*)d_in[52], (const float*)d_in[53],
            (const float*)d_in[54], (const float*)d_in[55], (void*)B, 16);
  }

  // linear: [16,128] f16 @ packed [128,256] + bias -> fp32 d_out [16,256]
  {
    const int tiles_ = 1 * (256 / 16);  // 16 tiles -> 2 blocks of 8 waves
    conv_wmma<128, 256, 1, 1, 1, 1, 0, 1, 1, true>
        <<<dim3((tiles_ + 7) / 8), dim3(256), 0, stream>>>(
            B, wp[11], (const float*)d_in[57], nullptr, nullptr, nullptr,
            d_out, 16);
  }

#undef RUN_CONV
#undef RUN_POOL
}
